// SelfAttention_56100862820423
// MI455X (gfx1250) — matrix operand order repaired and checked by reading, not compiled, not run on hardware
//
#include <hip/hip_runtime.h>
#include <hip/hip_bf16.h>

#define D_MODEL 1024
#define N_HEADS 16
#define HD      64
#define B_SZ    4
#define T_SEQ   2048
#define NTOK    (B_SZ * T_SEQ)   /* 8192 */
#define E3      (3 * D_MODEL)    /* 3072 */

typedef _Float16 v4h  __attribute__((ext_vector_type(4)));
typedef _Float16 v8h  __attribute__((ext_vector_type(8)));
typedef _Float16 v16h __attribute__((ext_vector_type(16)));
typedef float    v8f  __attribute__((ext_vector_type(8)));
typedef unsigned int u32x4 __attribute__((ext_vector_type(4)));
typedef int          i32x4 __attribute__((ext_vector_type(4)));
typedef int          i32x8 __attribute__((ext_vector_type(8)));

#if __has_builtin(__builtin_amdgcn_tensor_load_to_lds) && __has_builtin(__builtin_amdgcn_s_wait_tensorcnt)
#define USE_TDM 1
#else
#define USE_TDM 0
#endif

static __device__ __forceinline__ v8f vzero8() {
  v8f z;
#pragma unroll
  for (int i = 0; i < 8; ++i) z[i] = 0.0f;
  return z;
}

static __device__ __forceinline__ v8f wmma_f16(v16h a, v16h b, v8f c) {
  // D = A(16x32 f16) x B(32x16 f16) + C(16x16 f32)
  return __builtin_amdgcn_wmma_f32_16x16x32_f16(false, a, false, b,
                                                (short)0, c, false, false);
}

static __device__ __forceinline__ v16h join8(v8h lo, v8h hi) {
  return __builtin_shufflevector(lo, hi, 0, 1, 2, 3, 4, 5, 6, 7,
                                         8, 9, 10, 11, 12, 13, 14, 15);
}

// A fragment (16x32 f16) from row-major [row][k], leading dim ldk (halves).
// lane<16: row=lane, K 0..7 (v0..3) & 16..23 (v4..7); lane>=16: row=lane-16, K 8..15 & 24..31.
static __device__ __forceinline__ v16h load_fragA(const _Float16* base, int ldk, int lane) {
  int row = lane & 15, hi = lane >> 4;
  const _Float16* p = base + (size_t)row * ldk + hi * 8;
  return join8(*(const v8h*)p, *(const v8h*)(p + 16));
}

// B fragment (32x16 f16) with B[k][n] = S[n][k]; S row-major [n][k], leading dim ldk.
// lane<16: N=lane, K 0..7 (v0..3) & 16..23 (v4..7); lane>=16: N=lane-16, K 8..15 & 24..31.
static __device__ __forceinline__ v16h load_fragB(const _Float16* base, size_t ldk, int lane) {
  int n = lane & 15, hi = lane >> 4;
  const _Float16* p = base + (size_t)n * ldk + hi * 8;
  return join8(*(const v8h*)p, *(const v8h*)(p + 16));
}

#if USE_TDM
// 2D TDM descriptor: load tile_d1 rows x tile_d0 f16 elements (row stride in elements)
// from global into LDS at byte offset lds_byte_off (tile stored densely row-major).
static __device__ __forceinline__ void tdm_load_tile_f16(
    unsigned lds_byte_off, const _Float16* gsrc,
    unsigned tile_d0, unsigned tile_d1, unsigned row_stride) {
  unsigned long long ga = (unsigned long long)(uintptr_t)gsrc;
  u32x4 g0;
  g0[0] = 1u;                                              // count=1, user mode
  g0[1] = lds_byte_off;                                    // lds_addr (bytes)
  g0[2] = (unsigned)ga;                                    // global_addr[31:0]
  g0[3] = ((unsigned)(ga >> 32) & 0x01ffffffu) | (2u << 30); // [56:32] | type=2
  i32x8 g1;
  g1[0] = 0x10000;                          // data_size=2B; no mask/flags/pad
  g1[1] = (int)(tile_d0 << 16);             // tensor_dim0[15:0] (== tile_d0)
  g1[2] = (int)(tile_d1 << 16);             // tensor_dim0[31:16]=0 | tensor_dim1[15:0]
  g1[3] = (int)(tile_d0 << 16);             // tensor_dim1[31:16]=0 | tile_dim0
  g1[4] = (int)tile_d1;                     // tile_dim1 | tile_dim2=0
  g1[5] = (int)row_stride;                  // tensor_dim0_stride[31:0]
  g1[6] = 0;                                // stride[47:32]=0 | dim1_stride[15:0]=0
  g1[7] = 0;
  i32x4 z4 = {0, 0, 0, 0};
  i32x8 z8 = {0, 0, 0, 0, 0, 0, 0, 0};
  __builtin_amdgcn_tensor_load_to_lds(g0, g1, z4, z4, z8, 0);
}
#endif

// ---------------------------------------------------------------- convert f32->f16 (x4 vectorized)
__global__ __launch_bounds__(256) void cvt_f16_kernel(const float* __restrict__ src,
                                                      _Float16* __restrict__ dst, int n4) {
  int i = blockIdx.x * blockDim.x + threadIdx.x;
  int stride = gridDim.x * blockDim.x;
  for (; i < n4; i += stride) {
    float4 v = ((const float4*)src)[i];
    v4h h;
    h[0] = (_Float16)v.x; h[1] = (_Float16)v.y;
    h[2] = (_Float16)v.z; h[3] = (_Float16)v.w;
    ((v4h*)dst)[i] = h;
  }
}

// ---------------------------------------------------------------- QKV projection GEMM
// qkv[m][e] = sum_k xh[m][k] * wh[e][k] + bias[e]; wave tile 16(M) x 64(N), 8 waves in M.
// True ping-pong register double-buffering (k-loop unrolled x2, no rotation copies).
__global__ __launch_bounds__(256) void qkv_gemm_kernel(
    const _Float16* __restrict__ xh, const _Float16* __restrict__ wh,
    const float* __restrict__ bias,
    _Float16* __restrict__ qh, _Float16* __restrict__ kh, _Float16* __restrict__ vt) {
  int lane = threadIdx.x & 31;
  int w    = threadIdx.x >> 5;
  int m0   = blockIdx.y * 128 + w * 16;
  int n0   = blockIdx.x * 64;

  v8f c[4];
#pragma unroll
  for (int j = 0; j < 4; ++j) c[j] = vzero8();

  const _Float16* abase = xh + (size_t)m0 * D_MODEL;
  const _Float16* bbase = wh + (size_t)n0 * D_MODEL;

  v16h a0 = load_fragA(abase, D_MODEL, lane);
  v16h b0[4], a1, b1[4];
#pragma unroll
  for (int j = 0; j < 4; ++j)
    b0[j] = load_fragB(bbase + (size_t)(16 * j) * D_MODEL, D_MODEL, lane);

  for (int k0 = 0; k0 < D_MODEL; k0 += 64) {
    // prefetch k0+32 into set 1 while set 0 computes
    int k1 = k0 + 32;
    a1 = load_fragA(abase + k1, D_MODEL, lane);
#pragma unroll
    for (int j = 0; j < 4; ++j)
      b1[j] = load_fragB(bbase + (size_t)(16 * j) * D_MODEL + k1, D_MODEL, lane);
#pragma unroll
    for (int j = 0; j < 4; ++j) c[j] = wmma_f16(a0, b0[j], c[j]);

    // prefetch k0+64 into set 0 while set 1 computes (last iter: dummy reload of 0)
    int k2 = (k0 + 64 < D_MODEL) ? (k0 + 64) : 0;
    a0 = load_fragA(abase + k2, D_MODEL, lane);
#pragma unroll
    for (int j = 0; j < 4; ++j)
      b0[j] = load_fragB(bbase + (size_t)(16 * j) * D_MODEL + k2, D_MODEL, lane);
#pragma unroll
    for (int j = 0; j < 4; ++j) c[j] = wmma_f16(a1, b1[j], c[j]);
  }

  // Epilogue: C VGPR r -> (row=r+8*hi, col=lane&15); scatter f16 Q/K/Vt, fold bias + Q scale.
  int col = lane & 15, hi = lane >> 4;
  int bidx  = m0 >> 11;
  int tbase = m0 & (T_SEQ - 1);
#pragma unroll
  for (int j = 0; j < 4; ++j) {
    int e  = n0 + 16 * j + col;
    int which = e >> 10;            // 0=Q 1=K 2=V (uniform per 16-col tile)
    int h  = (e >> 6) & 15;
    int hd = e & 63;
    float bv  = bias[e];
    float scl = (which == 0) ? 0.125f : 1.0f;   // 1/sqrt(64) folded into Q
    size_t bh = (size_t)bidx * N_HEADS + h;
#pragma unroll
    for (int r = 0; r < 8; ++r) {
      int t = tbase + r + 8 * hi;
      _Float16 hv = (_Float16)((c[j][r] + bv) * scl);
      if (which == 0)      qh[(bh * T_SEQ + t) * HD + hd] = hv;
      else if (which == 1) kh[(bh * T_SEQ + t) * HD + hd] = hv;
      else                 vt[(bh * HD + hd) * T_SEQ + t] = hv;
    }
  }
}

// ---------------------------------------------------------------- flash attention
// Block = 8 waves = 128 query rows of one (b,h); all waves share K/V tiles.
// K (32x64) and Vt (64x32) tiles staged into double-buffered LDS by the TDM
// (wave 0 issues tensor_load_to_lds; s_wait_tensorcnt + barrier publishes).
__global__ __launch_bounds__(256) void attn_kernel(
    const _Float16* __restrict__ qh, const _Float16* __restrict__ kh,
    const _Float16* __restrict__ vt, float* __restrict__ out) {
  __shared__ __align__(16) _Float16 kbuf[2][32 * HD];   // [key][hd]
  __shared__ __align__(16) _Float16 vbuf[2][HD * 32];   // [hd][t]
  __shared__ __align__(16) _Float16 plds[8][16 * 32];   // per-wave P tile

  int lane = threadIdx.x & 31;
  int w    = threadIdx.x >> 5;
  int col  = lane & 15, hi = lane >> 4;
  int bh   = blockIdx.y;
  int q0   = blockIdx.x * 128 + w * 16;

  const _Float16* kh_bh = kh + (size_t)bh * T_SEQ * HD;
  const _Float16* vt_bh = vt + (size_t)bh * HD * T_SEQ;

  const _Float16* qbase = qh + ((size_t)bh * T_SEQ + q0) * HD;
  v16h aq0 = load_fragA(qbase,      HD, lane);
  v16h aq1 = load_fragA(qbase + 32, HD, lane);

  v8f o[4];
#pragma unroll
  for (int j = 0; j < 4; ++j) o[j] = vzero8();
  float m8[8], l8[8];
#pragma unroll
  for (int r = 0; r < 8; ++r) { m8[r] = -1.0e30f; l8[r] = 0.0f; }

  auto stage = [&](int buf, int kv) {
#if USE_TDM
    if (w == 0) {
      tdm_load_tile_f16((unsigned)(uintptr_t)&kbuf[buf][0],
                        kh_bh + (size_t)kv * HD, HD, 32, HD);
      tdm_load_tile_f16((unsigned)(uintptr_t)&vbuf[buf][0],
                        vt_bh + kv, 32, HD, T_SEQ);
    }
#else
    int tid = threadIdx.x;
    ((v8h*)&kbuf[buf][0])[tid] = ((const v8h*)(kh_bh + (size_t)kv * HD))[tid];
    int vrow = tid >> 2, vseg = (tid & 3) * 8;
    *(v8h*)&vbuf[buf][vrow * 32 + vseg] =
        *(const v8h*)(vt_bh + (size_t)vrow * T_SEQ + kv + vseg);
#endif
  };

  const int NSTEP = T_SEQ / 32;   // 64
  stage(0, 0);

  for (int i = 0; i < NSTEP; ++i) {
    int buf = i & 1;
    if (i + 1 < NSTEP) stage(buf ^ 1, (i + 1) * 32);   // DMA next tile behind compute
#if USE_TDM
    if (w == 0) {
      if (i + 1 < NSTEP) __builtin_amdgcn_s_wait_tensorcnt(2);  // oldest 2 (this buf) done
      else               __builtin_amdgcn_s_wait_tensorcnt(0);
    }
#endif
    __syncthreads();   // current buffer visible to all waves

    // --- scores S = Qs * K^T (two 16x16 tiles over 32 keys) ---
    const _Float16* kb = &kbuf[buf][0];
    v16h b00 = load_fragB(kb,                HD, lane);
    v16h b01 = load_fragB(kb + 32,           HD, lane);
    v16h b10 = load_fragB(kb + 16 * HD,      HD, lane);
    v16h b11 = load_fragB(kb + 16 * HD + 32, HD, lane);
    v8f s0 = vzero8(), s1 = vzero8();
    s0 = wmma_f16(aq0, b00, s0);
    s0 = wmma_f16(aq1, b01, s0);
    s1 = wmma_f16(aq0, b10, s1);
    s1 = wmma_f16(aq1, b11, s1);

    // --- online softmax (16-lane group == one tile row) ---
#pragma unroll
    for (int r = 0; r < 8; ++r) {
      float e0 = s0[r], e1 = s1[r];
      float mx = fmaxf(e0, e1);
#pragma unroll
      for (int off = 8; off > 0; off >>= 1) mx = fmaxf(mx, __shfl_xor(mx, off, 16));
      float mnew = fmaxf(m8[r], mx);
      float resc = __expf(m8[r] - mnew);
      float p0 = __expf(e0 - mnew);
      float p1 = __expf(e1 - mnew);
      float rs = p0 + p1;
#pragma unroll
      for (int off = 8; off > 0; off >>= 1) rs += __shfl_xor(rs, off, 16);
      l8[r] = l8[r] * resc + rs;
      m8[r] = mnew;
      o[0][r] *= resc; o[1][r] *= resc; o[2][r] *= resc; o[3][r] *= resc;
      int row = r + 8 * hi;
      plds[w][row * 32 + col]      = (_Float16)p0;
      plds[w][row * 32 + 16 + col] = (_Float16)p1;
    }
    __syncthreads();   // cross-lane P relayout visible

    // --- O += P * V ---
    v16h ap = load_fragA(&plds[w][0], 32, lane);
#pragma unroll
    for (int cc = 0; cc < 4; ++cc) {
      v16h bv = load_fragB(&vbuf[buf][(size_t)(cc * 16) * 32], 32, lane);
      o[cc] = wmma_f16(ap, bv, o[cc]);
    }
    __syncthreads();   // all reads of buf done before it is re-staged
  }

  // --- normalize and store [b, t, h*64+hd] fp32 ---
  int b = bh >> 4, h = bh & 15;
#pragma unroll
  for (int r = 0; r < 8; ++r) {
    float inv = 1.0f / l8[r];
    int row = r + 8 * hi;
    size_t obase = ((size_t)b * T_SEQ + q0 + row) * D_MODEL + (size_t)h * HD + col;
#pragma unroll
    for (int cc = 0; cc < 4; ++cc) out[obase + cc * 16] = o[cc][r] * inv;
  }
}

// ---------------------------------------------------------------- launch
extern "C" void kernel_launch(void* const* d_in, const int* in_sizes, int n_in,
                              void* d_out, int out_size, void* d_ws, size_t ws_size,
                              hipStream_t stream) {
  const float* x    = (const float*)d_in[0];  // [4,2048,1024]
  const float* Wq   = (const float*)d_in[1];  // [3072,1024]
  const float* bias = (const float*)d_in[2];  // [3072]
  float* out = (float*)d_out;                 // [4,2048,1024]

  char* ws = (char*)d_ws;
  const size_t SZ_XH = (size_t)NTOK * D_MODEL * sizeof(_Float16);
  const size_t SZ_WH = (size_t)E3 * D_MODEL * sizeof(_Float16);
  const size_t SZ_Q  = (size_t)NTOK * D_MODEL * sizeof(_Float16);

  _Float16* xh = (_Float16*)(ws);
  _Float16* wh = (_Float16*)(ws + SZ_XH);
  _Float16* qh = (_Float16*)(ws + SZ_XH + SZ_WH);
  _Float16* kh = (_Float16*)(ws + SZ_XH + SZ_WH + SZ_Q);
  _Float16* vt = (_Float16*)(ws + SZ_XH + SZ_WH + 2 * SZ_Q);

  cvt_f16_kernel<<<1024, 256, 0, stream>>>(x, xh, NTOK * D_MODEL / 4);
  cvt_f16_kernel<<<512, 256, 0, stream>>>(Wq, wh, E3 * D_MODEL / 4);

  dim3 g1(E3 / 64, NTOK / 128);  // (48, 64)
  qkv_gemm_kernel<<<g1, 256, 0, stream>>>(xh, wh, bias, qh, kh, vt);

  dim3 g2(T_SEQ / 128, B_SZ * N_HEADS);  // (16, 64)
  attn_kernel<<<g2, 256, 0, stream>>>(qh, kh, vt, out);
}
